// QuantLlamaModel_62251255988369
// MI455X (gfx1250) — 16-bit and 8-bit matrix operand order repaired and checked by reading, compiled, not run on hardware
//
#include <hip/hip_runtime.h>
#include <cstdint>
#include <cstddef>

#define Hdim   2048
#define NH     16
#define KVH    8
#define HD     128
#define NLAY   2
#define Iff    5632
#define Bsz    4
#define Slen   1024
#define Ttok   (Bsz * Slen)
#define QKV_N  (NH*HD + 2*KVH*HD)
#define GU_N   (2*Iff)
#define OLD    (NH*HD)

typedef _Float16 half_t;
typedef __attribute__((ext_vector_type(8)))  int      v8i;
typedef __attribute__((ext_vector_type(16))) _Float16 v16h;
typedef __attribute__((ext_vector_type(8)))  float    v8f;

__global__ __launch_bounds__(256) void wquant_kernel(
    const float* __restrict__ W, int8_t* __restrict__ Q,
    float* __restrict__ Sc, int K)
{
    int row = blockIdx.x, tid = threadIdx.x;
    const float* w = W + (size_t)row * K;
    __shared__ float red[256];
    float mx = 0.f;
    for (int j = tid; j < K; j += 256) mx = fmaxf(mx, fabsf(w[j]));
    red[tid] = mx; __syncthreads();
    for (int s = 128; s > 0; s >>= 1) {
        if (tid < s) red[tid] = fmaxf(red[tid], red[tid + s]);
        __syncthreads();
    }
    float sc = fmaxf(red[0] * (1.f / 127.f), 1e-8f);
    if (tid == 0) Sc[row] = sc;
    float rs = 1.f / sc;
    int8_t* q = Q + (size_t)row * K;
    for (int j = tid; j < K; j += 256)
        q[j] = (int8_t)(int)rintf(fminf(127.f, fmaxf(-127.f, w[j] * rs)));
}

__global__ __launch_bounds__(256) void rmsq_kernel(
    const float* __restrict__ X, const float* __restrict__ W,
    int8_t* __restrict__ Q, float* __restrict__ Sc)
{
    const int C = Hdim;
    int row = blockIdx.x, tid = threadIdx.x;
    const float* x = X + (size_t)row * C;
    float xv[8]; float ss = 0.f;
#pragma unroll
    for (int i = 0; i < 8; ++i) { xv[i] = x[tid + i * 256]; ss += xv[i] * xv[i]; }
    __shared__ float red[256];
    red[tid] = ss; __syncthreads();
    for (int s = 128; s > 0; s >>= 1) { if (tid < s) red[tid] += red[tid + s]; __syncthreads(); }
    float inv = rsqrtf(red[0] * (1.f / C) + 1e-6f);
    __syncthreads();
    float y[8]; float mx = 0.f;
#pragma unroll
    for (int i = 0; i < 8; ++i) {
        y[i] = xv[i] * inv * W[tid + i * 256];
        mx = fmaxf(mx, fabsf(y[i]));
    }
    red[tid] = mx; __syncthreads();
    for (int s = 128; s > 0; s >>= 1) { if (tid < s) red[tid] = fmaxf(red[tid], red[tid + s]); __syncthreads(); }
    float sc = fmaxf(red[0] * (1.f / 127.f), 1e-8f);
    if (tid == 0) Sc[row] = sc;
    float rs = 1.f / sc;
#pragma unroll
    for (int i = 0; i < 8; ++i)
        Q[(size_t)row * C + tid + i * 256] =
            (int8_t)(int)rintf(fminf(127.f, fmaxf(-127.f, y[i] * rs)));
}

__global__ __launch_bounds__(256) void quant_kernel(
    const float* __restrict__ X, int8_t* __restrict__ Q, float* __restrict__ Sc)
{
    const int C = OLD;
    int row = blockIdx.x, tid = threadIdx.x;
    const float* x = X + (size_t)row * C;
    float xv[8]; float mx = 0.f;
#pragma unroll
    for (int i = 0; i < 8; ++i) { xv[i] = x[tid + i * 256]; mx = fmaxf(mx, fabsf(xv[i])); }
    __shared__ float red[256];
    red[tid] = mx; __syncthreads();
    for (int s = 128; s > 0; s >>= 1) { if (tid < s) red[tid] = fmaxf(red[tid], red[tid + s]); __syncthreads(); }
    float sc = fmaxf(red[0] * (1.f / 127.f), 1e-8f);
    if (tid == 0) Sc[row] = sc;
    float rs = 1.f / sc;
#pragma unroll
    for (int i = 0; i < 8; ++i)
        Q[(size_t)row * C + tid + i * 256] =
            (int8_t)(int)rintf(fminf(127.f, fmaxf(-127.f, xv[i] * rs)));
}

__global__ __launch_bounds__(256) void actq_kernel(
    const float* __restrict__ GU, int8_t* __restrict__ Q, float* __restrict__ Sc)
{
    int row = blockIdx.x, tid = threadIdx.x;
    const float* g = GU + (size_t)row * GU_N;
    float a[22]; float mx = 0.f;
#pragma unroll
    for (int i = 0; i < 22; ++i) {
        int j = tid + i * 256;
        float gv = g[j], uv = g[Iff + j];
        float s = gv / (1.f + __expf(-gv));
        a[i] = s * uv;
        mx = fmaxf(mx, fabsf(a[i]));
    }
    __shared__ float red[256];
    red[tid] = mx; __syncthreads();
    for (int s = 128; s > 0; s >>= 1) { if (tid < s) red[tid] = fmaxf(red[tid], red[tid + s]); __syncthreads(); }
    float sc = fmaxf(red[0] * (1.f / 127.f), 1e-8f);
    if (tid == 0) Sc[row] = sc;
    float rs = 1.f / sc;
#pragma unroll
    for (int i = 0; i < 22; ++i)
        Q[(size_t)row * Iff + tid + i * 256] =
            (int8_t)(int)rintf(fminf(127.f, fmaxf(-127.f, a[i] * rs)));
}

__global__ __launch_bounds__(256) void final_rms_kernel(
    const float* __restrict__ X, const float* __restrict__ W, float* __restrict__ Out)
{
    const int C = Hdim;
    int row = blockIdx.x, tid = threadIdx.x;
    const float* x = X + (size_t)row * C;
    float xv[8]; float ss = 0.f;
#pragma unroll
    for (int i = 0; i < 8; ++i) { xv[i] = x[tid + i * 256]; ss += xv[i] * xv[i]; }
    __shared__ float red[256];
    red[tid] = ss; __syncthreads();
    for (int s = 128; s > 0; s >>= 1) { if (tid < s) red[tid] += red[tid + s]; __syncthreads(); }
    float inv = rsqrtf(red[0] * (1.f / C) + 1e-6f);
#pragma unroll
    for (int i = 0; i < 8; ++i)
        Out[(size_t)row * C + tid + i * 256] = xv[i] * inv * W[tid + i * 256];
}

__global__ __launch_bounds__(256) void gemm_i8_kernel(
    const int8_t* __restrict__ A, const float* __restrict__ As,
    const int8_t* __restrict__ Bw, const float* __restrict__ Bs,
    const float* __restrict__ Res, float* __restrict__ Out,
    int M, int N, int K, int add_res)
{
    int lane = threadIdx.x & 31;
    int wave = threadIdx.x >> 5;
    int wm = wave & 1;
    int wn = wave >> 1;
    int mbase = blockIdx.y * 64  + wm * 32;
    int nbase = blockIdx.x * 256 + wn * 64;
    int mlo = lane & 15, half = lane >> 4;

    v8i acc[2][4];
#pragma unroll
    for (int i = 0; i < 2; ++i)
#pragma unroll
        for (int j = 0; j < 4; ++j)
#pragma unroll
            for (int r = 0; r < 8; ++r) acc[i][j][r] = 0;

    const int8_t* arow = A  + (size_t)(mbase + mlo) * K + half * 16;
    const int8_t* brow = Bw + (size_t)(nbase + mlo) * K + half * 16;

    for (int k0 = 0; k0 < K; k0 += 64) {
        v8i af[2], bf[4];
#pragma unroll
        for (int i = 0; i < 2; ++i) {
            const int8_t* p = arow + (size_t)i * 16 * K + k0;
#pragma unroll
            for (int c = 0; c < 4; ++c) {
                int2 t = *(const int2*)(p + (c >> 1) * 32 + (c & 1) * 8);
                af[i][2 * c] = t.x; af[i][2 * c + 1] = t.y;
            }
        }
#pragma unroll
        for (int j = 0; j < 4; ++j) {
            const int8_t* p = brow + (size_t)j * 16 * K + k0;
            int4 t0 = *(const int4*)(p);
            int4 t1 = *(const int4*)(p + 32);
            bf[j][0] = t0.x; bf[j][1] = t0.y; bf[j][2] = t0.z; bf[j][3] = t0.w;
            bf[j][4] = t1.x; bf[j][5] = t1.y; bf[j][6] = t1.z; bf[j][7] = t1.w;
            __builtin_prefetch(p + 128, 0, 1);
        }
#pragma unroll
        for (int i = 0; i < 2; ++i)
#pragma unroll
            for (int j = 0; j < 4; ++j)
                acc[i][j] = __builtin_amdgcn_wmma_i32_16x16x64_iu8(
                    true, af[i], true, bf[j], acc[i][j], false, false);
    }

#pragma unroll
    for (int i = 0; i < 2; ++i)
#pragma unroll
        for (int j = 0; j < 4; ++j) {
            int col = nbase + j * 16 + mlo;
            float bs = Bs[col];
#pragma unroll
            for (int r = 0; r < 8; ++r) {
                int m = mbase + i * 16 + r + 8 * half;
                float v = (float)acc[i][j][r] * As[m] * bs;
                size_t idx = (size_t)m * N + col;
                Out[idx] = add_res ? (Res[idx] + v) : v;
            }
        }
}

__global__ __launch_bounds__(128) void rope_pack_kernel(
    const float* __restrict__ QKV, half_t* __restrict__ Q,
    half_t* __restrict__ Kc, half_t* __restrict__ Vt)
{
    int bid = blockIdx.x;
    int t = bid >> 5, hh = bid & 31;
    int d = threadIdx.x;
    int b = t / Slen, s = t % Slen;
    float ang = (float)s * __powf(10000.f, -(float)(d & 63) * (1.f / 64.f));
    float cs = cosf(ang), sn = sinf(ang);
    if (hh < NH) {
        const float* x = QKV + (size_t)t * QKV_N + hh * HD;
        float xv = x[d];
        float rot = (d < 64) ? -x[d + 64] : x[d - 64];
        Q[(((size_t)b * NH + hh) * Slen + s) * HD + d] = (half_t)(xv * cs + rot * sn);
    } else if (hh < NH + KVH) {
        int kh = hh - NH;
        const float* x = QKV + (size_t)t * QKV_N + NH * HD + kh * HD;
        float xv = x[d];
        float rot = (d < 64) ? -x[d + 64] : x[d - 64];
        Kc[(((size_t)b * KVH + kh) * Slen + s) * HD + d] = (half_t)(xv * cs + rot * sn);
    } else {
        int vh = hh - NH - KVH;
        Vt[(((size_t)b * KVH + vh) * HD + d) * Slen + s] =
            (half_t)QKV[(size_t)t * QKV_N + NH * HD + KVH * HD + vh * HD + d];
    }
}

__device__ __forceinline__ float rowmax16(float v) {
#pragma unroll
    for (int m = 1; m < 16; m <<= 1) v = fmaxf(v, __shfl_xor(v, m, 32));
    return v;
}
__device__ __forceinline__ float rowsum16(float v) {
#pragma unroll
    for (int m = 1; m < 16; m <<= 1) v += __shfl_xor(v, m, 32);
    return v;
}

__global__ __launch_bounds__(256) void attn_kernel(
    const half_t* __restrict__ Qm, const half_t* __restrict__ Km,
    const half_t* __restrict__ Vm, float* __restrict__ O)
{
    __shared__ half_t pbuf[8][16][32];
    int lane = threadIdx.x & 31, wv = threadIdx.x >> 5;
    int gt = blockIdx.x * 8 + wv;
    int qt = gt & 63;
    int h  = (gt >> 6) & 15;
    int b  = gt >> 10;
    int kh = h >> 1;
    const half_t* Qb  = Qm + (((size_t)b * NH  + h ) * Slen + qt * 16) * HD;
    const half_t* Kb  = Km + (((size_t)b * KVH + kh) * Slen) * HD;
    const half_t* Vtb = Vm + (((size_t)b * KVH + kh) * HD) * Slen;
    int mlo = lane & 15, half = lane >> 4;

    v16h qf[4];
#pragma unroll
    for (int c = 0; c < 4; ++c) {
        const half_t* p = Qb + (size_t)mlo * HD + c * 32 + half * 16;
#pragma unroll
        for (int j = 0; j < 8; ++j) { qf[c][j] = p[j]; qf[c][8 + j] = p[8 + j]; }
    }

    v8f oacc[8];
#pragma unroll
    for (int nb = 0; nb < 8; ++nb)
#pragma unroll
        for (int r = 0; r < 8; ++r) oacc[nb][r] = 0.f;
    float mstate[8], lstate[8];
#pragma unroll
    for (int r = 0; r < 8; ++r) { mstate[r] = -1e30f; lstate[r] = 0.f; }

    const float qscale = 0.08838834764831845f;
    int qmaxrow = qt * 16 + 15;

    for (int kc = 0; kc <= qmaxrow; kc += 32) {
        v8f s0, s1;
#pragma unroll
        for (int r = 0; r < 8; ++r) { s0[r] = 0.f; s1[r] = 0.f; }
#pragma unroll
        for (int c = 0; c < 4; ++c) {
            v16h k0f, k1f;
            const half_t* p0 = Kb + (size_t)(kc + mlo) * HD + c * 32 + half * 16;
            const half_t* p1 = p0 + (size_t)16 * HD;
#pragma unroll
            for (int j = 0; j < 16; ++j) { k0f[j] = p0[j]; k1f[j] = p1[j]; }
            s0 = __builtin_amdgcn_wmma_f32_16x16x32_f16(false, qf[c], false, k0f,
                                                        (short)0, s0, false, false);
            s1 = __builtin_amdgcn_wmma_f32_16x16x32_f16(false, qf[c], false, k1f,
                                                        (short)0, s1, false, false);
        }
#pragma unroll
        for (int r = 0; r < 8; ++r) {
            int qrow = qt * 16 + r + 8 * half;
            int c0 = kc + mlo, c1 = kc + 16 + mlo;
            float v0 = (c0 <= qrow) ? tanhf(s0[r] * qscale * 0.02f) * 50.f : -1e30f;
            float v1 = (c1 <= qrow) ? tanhf(s1[r] * qscale * 0.02f) * 50.f : -1e30f;
            float cm = rowmax16(fmaxf(v0, v1));
            float nm = fmaxf(mstate[r], cm);
            float al = __expf(mstate[r] - nm);
            mstate[r] = nm;
            float p0 = (c0 <= qrow) ? __expf(v0 - nm) : 0.f;
            float p1 = (c1 <= qrow) ? __expf(v1 - nm) : 0.f;
            lstate[r] = lstate[r] * al + rowsum16(p0 + p1);
#pragma unroll
            for (int nb = 0; nb < 8; ++nb) oacc[nb][r] *= al;
            pbuf[wv][r + 8 * half][mlo]      = (half_t)p0;
            pbuf[wv][r + 8 * half][16 + mlo] = (half_t)p1;
        }
        asm volatile("s_wait_dscnt 0" ::: "memory");
        v16h pf;
        {
            const half_t* rowp = &pbuf[wv][mlo][0];
            int ko = half * 16;
#pragma unroll
            for (int j = 0; j < 8; ++j) { pf[j] = rowp[ko + j]; pf[8 + j] = rowp[8 + ko + j]; }
        }
#pragma unroll
        for (int nb = 0; nb < 8; ++nb) {
            v16h vf;
            const half_t* vp = Vtb + (size_t)(nb * 16 + mlo) * Slen + kc + half * 16;
#pragma unroll
            for (int j = 0; j < 16; ++j) vf[j] = vp[j];
            oacc[nb] = __builtin_amdgcn_wmma_f32_16x16x32_f16(false, pf, false, vf,
                                                              (short)0, oacc[nb], false, false);
        }
    }

    int tbase = b * Slen + qt * 16;
#pragma unroll
    for (int r = 0; r < 8; ++r) {
        float invl = 1.f / lstate[r];
        int m = tbase + r + 8 * half;
#pragma unroll
        for (int nb = 0; nb < 8; ++nb)
            O[(size_t)m * OLD + h * HD + nb * 16 + mlo] = oacc[nb][r] * invl;
    }
}

extern "C" void kernel_launch(void* const* d_in, const int* in_sizes, int n_in,
                              void* d_out, int out_size, void* d_ws, size_t ws_size,
                              hipStream_t stream)
{
    (void)in_sizes; (void)n_in; (void)out_size; (void)ws_size;
    const float* emb   = (const float*)d_in[0];
    const float* ln1_w = (const float*)d_in[1];
    const float* ln2_w = (const float*)d_in[2];
    const float* wqkv  = (const float*)d_in[3];
    const float* wo    = (const float*)d_in[4];
    const float* wgu   = (const float*)d_in[5];
    const float* wd    = (const float*)d_in[6];
    const float* nw    = (const float*)d_in[7];
    float* out = (float*)d_out;

    char* ws = (char*)d_ws;
    size_t off = 0;
    auto take = [&](size_t n) -> char* {
        char* p = ws + off;
        off += (n + 255) & ~(size_t)255;
        return p;
    };
    float*   h      = (float*)  take((size_t)Ttok * Hdim * 4);
    int8_t*  qx     = (int8_t*) take((size_t)Ttok * Iff);
    float*   sx     = (float*)  take((size_t)Ttok * 4);
    float*   qkv    = (float*)  take((size_t)Ttok * QKV_N * 4);
    half_t*  q16    = (half_t*) take((size_t)Ttok * NH  * HD * 2);
    half_t*  k16    = (half_t*) take((size_t)Ttok * KVH * HD * 2);
    half_t*  v16t   = (half_t*) take((size_t)Ttok * KVH * HD * 2);
    float*   obuf   = (float*)  take((size_t)Ttok * OLD * 4);
    float*   gu     = (float*)  take((size_t)Ttok * GU_N * 4);
    int8_t*  wqkv_q = (int8_t*) take((size_t)NLAY * QKV_N * Hdim);
    float*   wqkv_s = (float*)  take((size_t)NLAY * QKV_N * 4);
    int8_t*  wo_q   = (int8_t*) take((size_t)NLAY * Hdim * OLD);
    float*   wo_s   = (float*)  take((size_t)NLAY * Hdim * 4);
    int8_t*  wgu_q  = (int8_t*) take((size_t)NLAY * GU_N * Hdim);
    float*   wgu_s  = (float*)  take((size_t)NLAY * GU_N * 4);
    int8_t*  wd_q   = (int8_t*) take((size_t)NLAY * Hdim * Iff);
    float*   wd_s   = (float*)  take((size_t)NLAY * Hdim * 4);

    hipMemcpyAsync(h, emb, (size_t)Ttok * Hdim * 4, hipMemcpyDeviceToDevice, stream);

    wquant_kernel<<<NLAY * QKV_N, 256, 0, stream>>>(wqkv, wqkv_q, wqkv_s, Hdim);
    wquant_kernel<<<NLAY * Hdim,  256, 0, stream>>>(wo,   wo_q,   wo_s,   OLD);
    wquant_kernel<<<NLAY * GU_N,  256, 0, stream>>>(wgu,  wgu_q,  wgu_s,  Hdim);
    wquant_kernel<<<NLAY * Hdim,  256, 0, stream>>>(wd,   wd_q,   wd_s,   Iff);

    for (int l = 0; l < NLAY; ++l) {
        rmsq_kernel<<<Ttok, 256, 0, stream>>>(h, ln1_w + (size_t)l * Hdim, qx, sx);
        gemm_i8_kernel<<<dim3(QKV_N / 256, Ttok / 64), 256, 0, stream>>>(
            qx, sx, wqkv_q + (size_t)l * QKV_N * Hdim, wqkv_s + (size_t)l * QKV_N,
            nullptr, qkv, Ttok, QKV_N, Hdim, 0);
        rope_pack_kernel<<<Ttok * 32, 128, 0, stream>>>(qkv, q16, k16, v16t);
        attn_kernel<<<(Bsz * NH * (Slen / 16)) / 8, 256, 0, stream>>>(q16, k16, v16t, obuf);
        quant_kernel<<<Ttok, 256, 0, stream>>>(obuf, qx, sx);
        gemm_i8_kernel<<<dim3(Hdim / 256, Ttok / 64), 256, 0, stream>>>(
            qx, sx, wo_q + (size_t)l * Hdim * OLD, wo_s + (size_t)l * Hdim,
            h, h, Ttok, Hdim, OLD, 1);
        rmsq_kernel<<<Ttok, 256, 0, stream>>>(h, ln2_w + (size_t)l * Hdim, qx, sx);
        gemm_i8_kernel<<<dim3(GU_N / 256, Ttok / 64), 256, 0, stream>>>(
            qx, sx, wgu_q + (size_t)l * GU_N * Hdim, wgu_s + (size_t)l * GU_N,
            nullptr, gu, Ttok, GU_N, Hdim, 0);
        actq_kernel<<<Ttok, 256, 0, stream>>>(gu, qx, sx);
        gemm_i8_kernel<<<dim3(Hdim / 256, Ttok / 64), 256, 0, stream>>>(
            qx, sx, wd_q + (size_t)l * Hdim * Iff, wd_s + (size_t)l * Hdim,
            h, h, Ttok, Hdim, Iff, 1);
    }
    final_rms_kernel<<<Ttok, 256, 0, stream>>>(h, nw, out);
}
